// MHA_85547158602000
// MI455X (gfx1250) — matrix operand order repaired and checked by reading, not compiled, not run on hardware
//
#include <hip/hip_runtime.h>

typedef __attribute__((ext_vector_type(16))) __bf16 v16bf;
typedef __attribute__((ext_vector_type(8)))  float  v8f;

#define BATCH 8
#define DIM 384
#define NTOK 1024
#define NHEAD 12
#define HDIM 32
#define QKVDIM 1152
#define SCALE 0.17677669529663687f

__device__ __forceinline__ unsigned short f32_to_bf16(float f) {
    unsigned int u = __float_as_uint(f);
    unsigned int r = (u + 0x7FFFu + ((u >> 16) & 1u)) >> 16;
    return (unsigned short)r;
}

__device__ __forceinline__ v8f vzero8() {
    v8f z = {0.f, 0.f, 0.f, 0.f, 0.f, 0.f, 0.f, 0.f};
    return z;
}

union FragU { v16bf v; uint4 q[2]; };
union AccU  { v8f v; float4 f[2]; };

// A fragment: 16x32 bf16 tile, row-major source, lane = row (&15),
// elements 0-7 -> K = g*8+0..7, elements 8-15 -> K = 16+g*8+0..7
__device__ __forceinline__ v16bf loadA(const unsigned short* base, int stride, int lane) {
    int r = lane & 15, g = lane >> 4;
    const unsigned short* p = base + (size_t)r * stride + g * 8;
    FragU u;
    u.q[0] = *(const uint4*)(p);
    u.q[1] = *(const uint4*)(p + 16);
    return u.v;
}

// B fragment: 32x16 bf16 (KxN). Source layout: element (k,n) at base + n*stride + k.
// lane = column n (&15), elements 0-7 -> K = g*8+0..7, elements 8-15 -> K = 16+g*8+0..7
__device__ __forceinline__ v16bf loadB(const unsigned short* base, int stride, int lane) {
    int n = lane & 15, g = lane >> 4;
    const unsigned short* p = base + (size_t)n * stride + g * 8;
    FragU u;
    u.q[0] = *(const uint4*)(p);
    u.q[1] = *(const uint4*)(p + 16);
    return u.v;
}

__device__ __forceinline__ v8f wmma_bf16(v16bf a, v16bf b, v8f c) {
    return __builtin_amdgcn_wmma_f32_16x16x32_bf16(false, a, false, b, (short)0, c, false, false);
}

// ---------------- prep kernels ----------------
__global__ __launch_bounds__(256) void cvt_bf16_kernel(const float* __restrict__ in,
                                                       unsigned short* __restrict__ out,
                                                       int nelem) {
    int i = blockIdx.x * blockDim.x + threadIdx.x;
    if (i < nelem) out[i] = f32_to_bf16(in[i]);
}

// x: (B, C, N) f32 -> xt: (B, N, C) bf16
__global__ __launch_bounds__(256) void transpose_x_kernel(const float* __restrict__ x,
                                                          unsigned short* __restrict__ xt) {
    int i = blockIdx.x * blockDim.x + threadIdx.x;
    if (i >= BATCH * DIM * NTOK) return;
    int n = i & (NTOK - 1);
    int t = i >> 10;
    int c = t % DIM;
    int b = t / DIM;
    xt[((size_t)b * NTOK + n) * DIM + c] = f32_to_bf16(x[i]);
}

// ---------------- QKV GEMM ----------------
// xt (B,N,C) bf16  @  W_qkv^T  -> q (B,H,N,d), k (B,H,N,d), vT (B,H,d,N), all bf16
__global__ __launch_bounds__(256) void qkv_gemm_kernel(const unsigned short* __restrict__ xt,
                                                       const unsigned short* __restrict__ Wq,
                                                       unsigned short* __restrict__ qb,
                                                       unsigned short* __restrict__ kb,
                                                       unsigned short* __restrict__ vtb) {
    const int NG = QKVDIM / 64;   // 18
    const int MG = NTOK / 32;     // 32
    int wid = (blockIdx.x * blockDim.x + threadIdx.x) >> 5;
    if (wid >= BATCH * MG * NG) return;
    int ng = wid % NG;
    int t2 = wid / NG;
    int mg = t2 % MG;
    int b  = t2 / MG;
    int lane = threadIdx.x & 31;
    int n = lane & 15, g = lane >> 4;
    int mbase = mg * 32, nbase = ng * 64;

    v8f acc[2][4];
    for (int m = 0; m < 2; ++m)
        for (int t = 0; t < 4; ++t) acc[m][t] = vzero8();

    const unsigned short* xrow = xt + ((size_t)b * NTOK + mbase) * DIM;
    for (int kk = 0; kk < DIM; kk += 32) {
        v16bf a0 = loadA(xrow + kk, DIM, lane);
        v16bf a1 = loadA(xrow + (size_t)16 * DIM + kk, DIM, lane);
        v16bf bf[4];
        for (int t = 0; t < 4; ++t)
            bf[t] = loadB(Wq + (size_t)(nbase + t * 16) * DIM + kk, DIM, lane);
        for (int t = 0; t < 4; ++t) {
            acc[0][t] = wmma_bf16(a0, bf[t], acc[0][t]);
            acc[1][t] = wmma_bf16(a1, bf[t], acc[1][t]);
        }
    }

    // Scatter D tiles: lane holds column o = nbase + t*16 + n, rows r+8g
    for (int m = 0; m < 2; ++m) {
        for (int t = 0; t < 4; ++t) {
            int o = nbase + t * 16 + n;
            int which = o / DIM;
            int rem = o - which * DIM;
            int head = rem >> 5;
            int dim = rem & 31;
            size_t bh = (size_t)b * NHEAD + head;
            for (int r = 0; r < 8; ++r) {
                int tok = mbase + m * 16 + r + 8 * g;
                unsigned short hv = f32_to_bf16(acc[m][t][r]);
                if (which == 0)      qb[(bh * NTOK + tok) * HDIM + dim] = hv;
                else if (which == 1) kb[(bh * NTOK + tok) * HDIM + dim] = hv;
                else                 vtb[(bh * HDIM + dim) * NTOK + tok] = hv;
            }
        }
    }
}

// ---------------- Flash attention ----------------
// per wave: 32 query rows of one (b, head); loop 32-key blocks with online softmax
__global__ __launch_bounds__(256) void flash_attn_kernel(const unsigned short* __restrict__ qb,
                                                         const unsigned short* __restrict__ kb,
                                                         const unsigned short* __restrict__ vtb,
                                                         unsigned short* __restrict__ ob) {
    __shared__ __align__(16) unsigned short pbuf[8][2][16 * 32];
    const int QG = NTOK / 32;  // 32
    int wid = (blockIdx.x * blockDim.x + threadIdx.x) >> 5;
    if (wid >= BATCH * NHEAD * QG) return;
    int qg = wid % QG;
    int t2 = wid / QG;
    int h = t2 % NHEAD;
    int b = t2 / NHEAD;
    int wslot = threadIdx.x >> 5;
    int lane = threadIdx.x & 31;
    int n = lane & 15, g = lane >> 4;

    size_t bh = (size_t)b * NHEAD + h;
    const unsigned short* qp = qb + bh * NTOK * HDIM;
    const unsigned short* kp = kb + bh * NTOK * HDIM;
    const unsigned short* vp = vtb + bh * HDIM * NTOK;
    int qbase = qg * 32;

    v16bf aq[2];
    aq[0] = loadA(qp + (size_t)qbase * HDIM, HDIM, lane);
    aq[1] = loadA(qp + (size_t)(qbase + 16) * HDIM, HDIM, lane);

    v8f o[2][2];
    float mrow[2][8], lrow[2][8];
    for (int m = 0; m < 2; ++m) {
        o[m][0] = vzero8();
        o[m][1] = vzero8();
        for (int r = 0; r < 8; ++r) { mrow[m][r] = -3.0e38f; lrow[m][r] = 0.f; }
    }

    for (int j = 0; j < NTOK; j += 32) {
        // S = (q * scale) . k^T  — B columns are key rows (contiguous in d)
        v16bf bk0 = loadB(kp + (size_t)j * HDIM, HDIM, lane);
        v16bf bk1 = loadB(kp + (size_t)(j + 16) * HDIM, HDIM, lane);
        v8f s[2][2];
        for (int m = 0; m < 2; ++m) {
            s[m][0] = wmma_bf16(aq[m], bk0, vzero8());
            s[m][1] = wmma_bf16(aq[m], bk1, vzero8());
        }
        // online softmax per M-tile; each row M = r+8g lives on 16 lanes of group g
        for (int m = 0; m < 2; ++m) {
            for (int r = 0; r < 8; ++r) {
                float v0 = s[m][0][r] * SCALE;
                float v1 = s[m][1][r] * SCALE;
                float mx = fmaxf(v0, v1);
                mx = fmaxf(mx, __shfl_xor(mx, 1));
                mx = fmaxf(mx, __shfl_xor(mx, 2));
                mx = fmaxf(mx, __shfl_xor(mx, 4));
                mx = fmaxf(mx, __shfl_xor(mx, 8));
                float mnew = fmaxf(mrow[m][r], mx);
                float alpha = __expf(mrow[m][r] - mnew);
                mrow[m][r] = mnew;
                float p0 = __expf(v0 - mnew);
                float p1 = __expf(v1 - mnew);
                s[m][0][r] = p0;
                s[m][1][r] = p1;
                float rs = p0 + p1;
                rs += __shfl_xor(rs, 1);
                rs += __shfl_xor(rs, 2);
                rs += __shfl_xor(rs, 4);
                rs += __shfl_xor(rs, 8);
                lrow[m][r] = lrow[m][r] * alpha + rs;
                o[m][0][r] *= alpha;
                o[m][1][r] *= alpha;
            }
            // P (D-layout) -> LDS bf16 row-major [16][32]
            for (int t = 0; t < 2; ++t)
                for (int r = 0; r < 8; ++r)
                    pbuf[wslot][m][(r + 8 * g) * 32 + t * 16 + n] = f32_to_bf16(s[m][t][r]);
        }
        asm volatile("s_wait_dscnt 0" ::: "memory");
        // P as A-fragments; V^T columns contiguous in key dim
        v16bf ap0 = loadA(&pbuf[wslot][0][0], 32, lane);
        v16bf ap1 = loadA(&pbuf[wslot][1][0], 32, lane);
        v16bf bv0 = loadB(vp + j, NTOK, lane);
        v16bf bv1 = loadB(vp + (size_t)16 * NTOK + j, NTOK, lane);
        o[0][0] = wmma_bf16(ap0, bv0, o[0][0]);
        o[0][1] = wmma_bf16(ap0, bv1, o[0][1]);
        o[1][0] = wmma_bf16(ap1, bv0, o[1][0]);
        o[1][1] = wmma_bf16(ap1, bv1, o[1][1]);
    }

    // normalize + store to (B, N, C) bf16
    for (int m = 0; m < 2; ++m) {
        for (int r = 0; r < 8; ++r) {
            float inv = 1.0f / lrow[m][r];
            o[m][0][r] *= inv;
            o[m][1][r] *= inv;
        }
        for (int t = 0; t < 2; ++t)
            for (int r = 0; r < 8; ++r) {
                int tok = qbase + m * 16 + r + 8 * g;
                ob[((size_t)b * NTOK + tok) * DIM + h * HDIM + t * 16 + n] =
                    f32_to_bf16(o[m][t][r]);
            }
    }
}

// ---------------- Projection GEMM ----------------
// attn_out (B,N,C) bf16 @ W_proj^T + b -> d_out (B,C,N) f32
__global__ __launch_bounds__(256) void proj_gemm_kernel(const unsigned short* __restrict__ ob,
                                                        const unsigned short* __restrict__ Wp,
                                                        const float* __restrict__ bias,
                                                        float* __restrict__ out) {
    const int NG = DIM / 64;    // 6
    const int MG = NTOK / 32;   // 32
    int wid = (blockIdx.x * blockDim.x + threadIdx.x) >> 5;
    if (wid >= BATCH * MG * NG) return;
    int ng = wid % NG;
    int t2 = wid / NG;
    int mg = t2 % MG;
    int b  = t2 / MG;
    int lane = threadIdx.x & 31;
    int n = lane & 15, g = lane >> 4;
    int mbase = mg * 32, nbase = ng * 64;

    v8f acc[2][4];
    for (int m = 0; m < 2; ++m)
        for (int t = 0; t < 4; ++t) acc[m][t] = vzero8();

    const unsigned short* arow = ob + ((size_t)b * NTOK + mbase) * DIM;
    for (int kk = 0; kk < DIM; kk += 32) {
        v16bf a0 = loadA(arow + kk, DIM, lane);
        v16bf a1 = loadA(arow + (size_t)16 * DIM + kk, DIM, lane);
        v16bf bf[4];
        for (int t = 0; t < 4; ++t)
            bf[t] = loadB(Wp + (size_t)(nbase + t * 16) * DIM + kk, DIM, lane);
        for (int t = 0; t < 4; ++t) {
            acc[0][t] = wmma_bf16(a0, bf[t], acc[0][t]);
            acc[1][t] = wmma_bf16(a1, bf[t], acc[1][t]);
        }
    }

    // bias + store: lane column = out channel; 8 rows per lane are consecutive tokens
    for (int t = 0; t < 4; ++t) {
        int ch = nbase + t * 16 + n;
        float bv = bias[ch];
        for (int m = 0; m < 2; ++m) {
            AccU u;
            u.v = acc[m][t];
            for (int r = 0; r < 8; ++r) u.v[r] += bv;
            float* dst = out + ((size_t)b * DIM + ch) * NTOK + mbase + m * 16 + 8 * g;
            *(float4*)(dst)     = u.f[0];
            *(float4*)(dst + 4) = u.f[1];
        }
    }
}

extern "C" void kernel_launch(void* const* d_in, const int* in_sizes, int n_in,
                              void* d_out, int out_size, void* d_ws, size_t ws_size,
                              hipStream_t stream) {
    const float* x     = (const float*)d_in[0];  // (8,384,32,32)
    const float* Wqkv  = (const float*)d_in[1];  // (1152,384)
    const float* Wproj = (const float*)d_in[2];  // (384,384)
    const float* bproj = (const float*)d_in[3];  // (384,)
    float* out = (float*)d_out;

    const size_t N_WQ = (size_t)QKVDIM * DIM;          // 442368
    const size_t N_WP = (size_t)DIM * DIM;             // 147456
    const size_t N_XT = (size_t)BATCH * NTOK * DIM;    // 3145728
    const size_t N_QK = (size_t)BATCH * NHEAD * NTOK * HDIM;  // 3145728

    char* ws = (char*)d_ws;
    size_t off = 0;
    auto carve = [&](size_t elems) {
        unsigned short* p = (unsigned short*)(ws + off);
        off += (elems * sizeof(unsigned short) + 255) & ~(size_t)255;
        return p;
    };
    unsigned short* wq_bf = carve(N_WQ);
    unsigned short* wp_bf = carve(N_WP);
    unsigned short* xt    = carve(N_XT);
    unsigned short* qbuf  = carve(N_QK);
    unsigned short* kbuf  = carve(N_QK);
    unsigned short* vtbuf = carve(N_QK);
    unsigned short* aobuf = carve(N_XT);

    const int T = 256;
    cvt_bf16_kernel<<<(int)((N_WQ + T - 1) / T), T, 0, stream>>>(Wqkv, wq_bf, (int)N_WQ);
    cvt_bf16_kernel<<<(int)((N_WP + T - 1) / T), T, 0, stream>>>(Wproj, wp_bf, (int)N_WP);
    transpose_x_kernel<<<(int)((N_XT + T - 1) / T), T, 0, stream>>>(x, xt);

    int qkv_waves = BATCH * (NTOK / 32) * (QKVDIM / 64);   // 4608
    qkv_gemm_kernel<<<(qkv_waves * 32 + T - 1) / T, T, 0, stream>>>(xt, wq_bf, qbuf, kbuf, vtbuf);

    int fa_waves = BATCH * NHEAD * (NTOK / 32);            // 3072
    flash_attn_kernel<<<(fa_waves * 32 + T - 1) / T, T, 0, stream>>>(qbuf, kbuf, vtbuf, aobuf);

    int pj_waves = BATCH * (NTOK / 32) * (DIM / 64);       // 1536
    proj_gemm_kernel<<<(pj_waves * 32 + T - 1) / T, T, 0, stream>>>(aobuf, wp_bf, bproj, out);
}
